// LocationSensitiveAttention_29807073034307
// MI455X (gfx1250) — matrix operand order repaired and checked by reading, not compiled, not run on hardware
//
#include <hip/hip_runtime.h>
#include <hip/hip_bf16.h>
#include <math.h>

// ---------------------------------------------------------------------------
// LocationSensitiveAttention for MI455X (gfx1250, wave32, WMMA + async-LDS)
// B=64, T=2000, E=512, D=1024, A=256, NF=32, K=31
// ---------------------------------------------------------------------------

#define BB  64
#define TT  2000
#define EE  512
#define DD  1024
#define AA  256
#define NF  32
#define KC  31      // conv kernel size
#define PAD 15
#define NKT 17      // K tiles: 16 x 32 for E=512, +1 for the location term

typedef __attribute__((ext_vector_type(16))) __bf16 v16bf;
typedef __attribute__((ext_vector_type(8)))  float  v8f;

union FragBF {
    v16bf v;
    unsigned u[8];
    uint4 q[2];
};

__device__ __forceinline__ unsigned short f2bf(float f) {
    unsigned u = __builtin_bit_cast(unsigned, f);
    unsigned r = u + 0x7FFFu + ((u >> 16) & 1u);   // round-to-nearest-even
    return (unsigned short)(r >> 16);
}

// ---------------------------------------------------------------------------
// Kernel 1: dec_proj[b,a] = sum_d decoder_state[b,d] * W_dec[d,a]
// ---------------------------------------------------------------------------
__global__ void k_dec_proj(const float* __restrict__ dec,
                           const float* __restrict__ Wdec,
                           float* __restrict__ out) {
    const int b = blockIdx.x;
    const int a = threadIdx.x;          // 256 threads == AA
    float s = 0.0f;
    const float* drow = dec + (size_t)b * DD;
    for (int d = 0; d < DD; ++d)
        s = fmaf(drow[d], Wdec[(size_t)d * AA + a], s);
    out[(size_t)b * AA + a] = s;
}

// ---------------------------------------------------------------------------
// Kernel 2: Mloc[k,a] = sum_f conv_w[f,0,k] * W_loc[f,a]   (k=31 row -> 0)
// ---------------------------------------------------------------------------
__global__ void k_mloc(const float* __restrict__ convw,
                       const float* __restrict__ Wloc,
                       float* __restrict__ Mloc) {
    const int k = blockIdx.x;           // 0..31
    const int a = threadIdx.x;          // 0..255
    float s = 0.0f;
    if (k < KC) {
        for (int f = 0; f < NF; ++f)
            s = fmaf(convw[f * KC + k], Wloc[f * AA + a], s);
    }
    Mloc[k * AA + a] = s;
}

// ---------------------------------------------------------------------------
// Kernel 3: pack [W_enc ; Mloc] -> bf16, transposed & k-tiled:
//   out[((kt*256 + n)*32 + k)] = bf16( K<512 ? W_enc[K,n] : Mloc[K-512,n] ),
// K = kt*32+k.  Each 16KB tile is contiguous -> perfect async staging.
// ---------------------------------------------------------------------------
__global__ void k_wpack(const float* __restrict__ Wenc,
                        const float* __restrict__ Mloc,
                        unsigned short* __restrict__ out) {
    const int idx = blockIdx.x * 256 + threadIdx.x;   // < 17*8192
    const int kt  = idx >> 13;
    const int rem = idx & 8191;
    const int n   = rem >> 5;
    const int k   = rem & 31;
    const int K   = kt * 32 + k;
    float v = (K < EE) ? Wenc[(size_t)K * AA + n] : Mloc[(K - EE) * AA + n];
    out[idx] = f2bf(v);
}

// ---------------------------------------------------------------------------
// Kernel 4: energy[b,t] = v . tanh(enc[b,t,:]@W_enc + dec_proj[b,:] + locA)
// WMMA bf16 GEMM: 64 t-rows x 256 cols per block, 17 K-steps of 32.
// B tiles staged via global_load_async_to_lds_b128 (triple buffered,
// s_wait_asynccnt), A tiles double buffered -> ONE barrier per K-step.
// B fragments register-pipelined so ds_load_b128 overlaps the XDL pipe
// (partial s_wait_dscnt instead of full drains).
// ---------------------------------------------------------------------------
#define PITCH  40                 // bf16 elements per LDS row
#define BSZ    (256 * PITCH * 2)  // bytes per B buffer

__global__ void __launch_bounds__(256)
k_energy(const float* __restrict__ enc,              // [B,T,E]
         const unsigned short* __restrict__ wpack,   // [17][256][32] bf16
         const float* __restrict__ prev,             // [B,T]
         const float* __restrict__ decp,             // [B,A]
         const float* __restrict__ vvec,             // [A]
         float* __restrict__ energy)                 // [B,T]
{
    __shared__ alignas(16) unsigned short ldsA[2][64 * PITCH];
    __shared__ alignas(16) unsigned short ldsB[3][256 * PITCH];
    __shared__ float ldsE[2 * 64];

    const int b   = blockIdx.y;
    const int t0  = blockIdx.x * 64;
    const int tid = threadIdx.x;
    const int w    = tid >> 5;
    const int lane = tid & 31;
    const int half = lane >> 4;
    const int l16  = lane & 15;
    const int mTile = w & 3;      // rows [mTile*16, +16)
    const int nHalf = w >> 2;     // n-tiles [nHalf*8, +8)

    // per-lane async-copy addressing: lane copies row n=tid, 4 x 16B chunks
    const unsigned long long gW = (unsigned long long)(const char*)wpack;
    const unsigned ldsBbase = (unsigned)(size_t)&ldsB[0][0];

    // ---- prologue: async-stage B tile 0 into buffer 0
    {
        unsigned long long ga = gW + (unsigned long long)(tid * 64);
        unsigned la = ldsBbase + (unsigned)(tid * (PITCH * 2));
        #pragma unroll
        for (int j = 0; j < 4; ++j) {
            asm volatile("global_load_async_to_lds_b128 %0, %1, off"
                         :: "v"(la + j * 16), "v"(ga + (unsigned long long)(j * 16))
                         : "memory");
        }
    }

    v8f acc[8];
    #pragma unroll
    for (int i = 0; i < 8; ++i) acc[i] = (v8f){0,0,0,0,0,0,0,0};

    int cur = 0;   // kk % 3
    for (int kk = 0; kk < NKT; ++kk) {
        const int nxt = (cur == 2) ? 0 : cur + 1;

        // ---- issue async copy for next B tile into buffer (kk+1)%3
        if (kk < NKT - 1) {
            unsigned long long ga = gW + (unsigned long long)(kk + 1) * 16384ull
                                       + (unsigned long long)(tid * 64);
            unsigned la = ldsBbase + (unsigned)nxt * (unsigned)BSZ
                                   + (unsigned)(tid * (PITCH * 2));
            #pragma unroll
            for (int j = 0; j < 4; ++j) {
                asm volatile("global_load_async_to_lds_b128 %0, %1, off"
                             :: "v"(la + j * 16), "v"(ga + (unsigned long long)(j * 16))
                             : "memory");
            }
        }

        // ---- stage A tile (64 rows x 32 k) fp32 -> bf16 into ldsA[kk&1]
        {
            unsigned short* bufA = &ldsA[kk & 1][0];
            const int c = (tid & 15) * 2;
            #pragma unroll
            for (int i = 0; i < 4; ++i) {
                const int r = (tid >> 4) + i * 16;
                const int t = t0 + r;
                float f0 = 0.0f, f1 = 0.0f;
                if (kk < NKT - 1) {
                    if (t < TT) {
                        const float2 e2 = *reinterpret_cast<const float2*>(
                            &enc[((size_t)b * TT + t) * EE + kk * 32 + c]);
                        f0 = e2.x; f1 = e2.y;
                    }
                } else {
                    // location chunk: sliding windows of prev attention
                    if (t < TT) {
                        int tp0 = t + c - PAD, tp1 = t + c + 1 - PAD;
                        if (c < KC && tp0 >= 0 && tp0 < TT)
                            f0 = prev[(size_t)b * TT + tp0];
                        if (c + 1 < KC && tp1 >= 0 && tp1 < TT)
                            f1 = prev[(size_t)b * TT + tp1];
                    }
                }
                unsigned packed = (unsigned)f2bf(f0) | ((unsigned)f2bf(f1) << 16);
                *reinterpret_cast<unsigned*>(&bufA[r * PITCH + c]) = packed;
            }
        }

        // ---- wait for this iteration's B tile (issued last iter) to land
        if (kk < NKT - 1) {
            asm volatile("s_wait_asynccnt 0x4" ::: "memory");
        } else {
            asm volatile("s_wait_asynccnt 0x0" ::: "memory");
        }
        __syncthreads();   // single barrier per K-step (A dbl-, B triple-buffered)

        const unsigned short* bufA = &ldsA[kk & 1][0];
        const unsigned short* bufB = &ldsB[cur][0];

        // ---- A fragment: two contiguous 16B runs per lane (k = 16*half + s, same order as B)
        FragBF af;
        {
            const int rowA = mTile * 16 + l16;
            af.q[0] = *reinterpret_cast<const uint4*>(&bufA[rowA * PITCH + 16 * half]);
            af.q[1] = *reinterpret_cast<const uint4*>(&bufA[rowA * PITCH + 16 * half + 8]);
        }

        // ---- 8 N-tiles of WMMA; B fragments register-pipelined
        FragBF bcur;
        {
            const int n0 = (nHalf * 8) * 16 + l16;
            bcur.q[0] = *reinterpret_cast<const uint4*>(&bufB[n0 * PITCH + 16 * half]);
            bcur.q[1] = *reinterpret_cast<const uint4*>(&bufB[n0 * PITCH + 16 * half + 8]);
        }
        #pragma unroll
        for (int nt = 0; nt < 8; ++nt) {
            FragBF bnxt;
            if (nt < 7) {
                const int n2 = (nHalf * 8 + nt + 1) * 16 + l16;
                bnxt.q[0] = *reinterpret_cast<const uint4*>(&bufB[n2 * PITCH + 16 * half]);
                bnxt.q[1] = *reinterpret_cast<const uint4*>(&bufB[n2 * PITCH + 16 * half + 8]);
            }
            acc[nt] = __builtin_amdgcn_wmma_f32_16x16x32_bf16(
                false, af.v, false, bcur.v, (short)0, acc[nt], false, false);
            if (nt < 7) bcur = bnxt;
        }

        cur = nxt;
    }

    // ---- epilogue: + dec_proj, tanh, dot with v, row-reduce
    float partial[8];
    #pragma unroll
    for (int r = 0; r < 8; ++r) partial[r] = 0.0f;

    #pragma unroll
    for (int nt = 0; nt < 8; ++nt) {
        const int col = (nHalf * 8 + nt) * 16 + l16;
        const float dp = decp[(size_t)b * AA + col];
        const float vw = vvec[col];
        #pragma unroll
        for (int r = 0; r < 8; ++r) {
            float pre = acc[nt][r] + dp;
            partial[r] += tanhf(pre) * vw;
        }
    }

    #pragma unroll
    for (int r = 0; r < 8; ++r) {
        float s = partial[r];
        #pragma unroll
        for (int m = 1; m < 16; m <<= 1)
            s += __shfl_xor(s, m, 32);
        if (l16 == 0)
            ldsE[nHalf * 64 + mTile * 16 + half * 8 + r] = s;
    }
    __syncthreads();

    if (tid < 64) {
        int t = t0 + tid;
        if (t < TT)
            energy[(size_t)b * TT + t] = ldsE[tid] + ldsE[64 + tid];
    }
}

// ---------------------------------------------------------------------------
// Kernel 5: masked softmax over T per batch row
// ---------------------------------------------------------------------------
__global__ void k_softmax(const float* __restrict__ energy,
                          const int* __restrict__ mask,
                          float* __restrict__ attn) {
    const int b = blockIdx.x;
    const int tid = threadIdx.x;
    __shared__ float red[256];

    float mx = -3.0e38f;
    for (int t = tid; t < TT; t += 256) {
        float e = (mask[(size_t)b * TT + t] == 0) ? -1e9f : energy[(size_t)b * TT + t];
        mx = fmaxf(mx, e);
    }
    red[tid] = mx; __syncthreads();
    for (int s = 128; s > 0; s >>= 1) {
        if (tid < s) red[tid] = fmaxf(red[tid], red[tid + s]);
        __syncthreads();
    }
    const float M = red[0];
    __syncthreads();

    float sum = 0.0f;
    for (int t = tid; t < TT; t += 256) {
        float e = (mask[(size_t)b * TT + t] == 0) ? -1e9f : energy[(size_t)b * TT + t];
        sum += expf(e - M);
    }
    red[tid] = sum; __syncthreads();
    for (int s = 128; s > 0; s >>= 1) {
        if (tid < s) red[tid] += red[tid + s];
        __syncthreads();
    }
    const float inv = 1.0f / red[0];

    for (int t = tid; t < TT; t += 256) {
        float e = (mask[(size_t)b * TT + t] == 0) ? -1e9f : energy[(size_t)b * TT + t];
        attn[(size_t)b * TT + t] = expf(e - M) * inv;
    }
}

// ---------------------------------------------------------------------------
// Kernel 6: context partials: part[tc,b,e] = sum_{t in chunk} attn[b,t]*enc[b,t,e]
// ---------------------------------------------------------------------------
#define TCHUNK 250
__global__ void k_ctx_part(const float* __restrict__ enc,
                           const float* __restrict__ attn,
                           float* __restrict__ part) {
    const int tc = blockIdx.x;   // 0..7
    const int b  = blockIdx.y;
    const int tid = threadIdx.x;
    __shared__ float wA[TCHUNK];
    const int tstart = tc * TCHUNK;
    if (tid < TCHUNK) wA[tid] = attn[(size_t)b * TT + tstart + tid];
    __syncthreads();

    float a0 = 0.0f, a1 = 0.0f;
    const float* ebase = enc + ((size_t)b * TT + tstart) * EE;
    for (int i = 0; i < TCHUNK; ++i) {
        const float wv = wA[i];
        a0 = fmaf(wv, ebase[(size_t)i * EE + tid],       a0);
        a1 = fmaf(wv, ebase[(size_t)i * EE + tid + 256], a1);
    }
    float* p = part + ((size_t)(tc * BB + b)) * EE;
    p[tid]       = a0;
    p[tid + 256] = a1;
}

// ---------------------------------------------------------------------------
// Kernel 7: deterministic reduce of the 8 partials -> context [B,E]
// ---------------------------------------------------------------------------
__global__ void k_ctx_reduce(const float* __restrict__ part,
                             float* __restrict__ ctx) {
    const int idx = blockIdx.x * 256 + threadIdx.x;  // 0..32767
    const int b = idx >> 9;
    const int e = idx & 511;
    float s = 0.0f;
    #pragma unroll
    for (int tc = 0; tc < 8; ++tc)
        s += part[((size_t)(tc * BB + b)) * EE + e];
    ctx[idx] = s;
}

// ---------------------------------------------------------------------------
extern "C" void kernel_launch(void* const* d_in, const int* in_sizes, int n_in,
                              void* d_out, int out_size, void* d_ws, size_t ws_size,
                              hipStream_t stream) {
    (void)in_sizes; (void)n_in; (void)out_size; (void)ws_size;

    const float* dec   = (const float*)d_in[0];  // [B,D]
    const float* enc   = (const float*)d_in[1];  // [B,T,E]
    const float* prev  = (const float*)d_in[2];  // [B,T]
    const int*   mask  = (const int*)  d_in[3];  // [B,T]
    const float* convw = (const float*)d_in[4];  // [NF,1,K]
    const float* Wenc  = (const float*)d_in[5];  // [E,A]
    const float* Wdec  = (const float*)d_in[6];  // [D,A]
    const float* Wloc  = (const float*)d_in[7];  // [NF,A]
    const float* vvec  = (const float*)d_in[8];  // [A]

    float* out_ctx  = (float*)d_out;                    // [B,E] = 32768
    float* out_attn = (float*)d_out + (size_t)BB * EE;  // [B,T] = 128000

    float* ws        = (float*)d_ws;
    float* wsDec     = ws;                        // B*A      = 16384 floats
    float* wsMloc    = ws + 16384;                // 32*A     =  8192
    float* wsEnergy  = ws + 24576;                // B*T      = 128000
    float* wsPart    = ws + 152576;               // 8*B*E    = 262144
    unsigned short* wsWpack = (unsigned short*)(ws + 414720); // 17*8192 bf16 (16B-aligned)

    k_dec_proj<<<BB, AA, 0, stream>>>(dec, Wdec, wsDec);
    k_mloc<<<32, AA, 0, stream>>>(convw, Wloc, wsMloc);
    k_wpack<<<(NKT * 8192) / 256, 256, 0, stream>>>(Wenc, wsMloc, wsWpack);

    dim3 gE(32, BB);   // 32 t-tiles of 64 rows, 64 batches
    k_energy<<<gE, 256, 0, stream>>>(enc, wsWpack, prev, wsDec, vvec, wsEnergy);

    k_softmax<<<BB, 256, 0, stream>>>(wsEnergy, mask, out_attn);

    dim3 gC(8, BB);
    k_ctx_part<<<gC, 256, 0, stream>>>(enc, out_attn, wsPart);
    k_ctx_reduce<<<128, 256, 0, stream>>>(wsPart, out_ctx);
}
